// HypergraphTransformerLayer_1271310319701
// MI455X (gfx1250) — compile-verified, matrix operand order checked by reading, not run on hardware
//
#include <hip/hip_runtime.h>
#include <math.h>

// ---------------------------------------------------------------------------
// HypergraphTransformerLayer for MI455X (gfx1250, wave32, WMMA)
// B=4, Q=K=4096, H=256 (8 heads x 32), E=4096, F=16
// GEMMs use v_wmma_f32_16x16x32_f16 (f16 A/B, f32 accumulate), K fully
// unrolled (8 wmma / tile). Weights pre-transposed+converted to f16 once.
// ---------------------------------------------------------------------------

#define B_   4
#define Q_   4096
#define K_   4096
#define H_   256
#define E_   4096
#define F_   16
#define NH_  8
#define HD_  32          // head dim
#define M_   (B_ * Q_)   // 16384 rows for node GEMMs

typedef _Float16 v16h __attribute__((ext_vector_type(16)));
typedef _Float16 v8h  __attribute__((ext_vector_type(8)));
typedef float    v8f  __attribute__((ext_vector_type(8)));

// ---------------------------------------------------------------------------
// Zero-fill (upd accumulator must start at 0 each call; harness poisons ws)
// ---------------------------------------------------------------------------
__global__ void zero_f32_kernel(float* __restrict__ p, int n4) {
    int i = blockIdx.x * blockDim.x + threadIdx.x;
    if (i < n4) ((float4*)p)[i] = make_float4(0.f, 0.f, 0.f, 0.f);
}

// ---------------------------------------------------------------------------
// Wt[n][k] = (f16) W[k][n]   (256x256; run once per weight matrix)
// ---------------------------------------------------------------------------
__global__ __launch_bounds__(256)
void transpose_cvt_kernel(const float* __restrict__ W, _Float16* __restrict__ Wt) {
    const int idx = blockIdx.x * 256 + threadIdx.x;   // 65536 elements
    const int n = idx >> 8;
    const int k = idx & 255;
    Wt[idx] = (_Float16)W[k * H_ + n];
}

// ---------------------------------------------------------------------------
// GEMM: Out[M,256] = act( A[M,256] @ W[256,256] + bias )
//   v_wmma_f32_16x16x32_f16; W given pre-transposed as f16 Wt[n][k].
//   block = 256 threads = 8 waves; wave tile 16x16; block tile 32(M) x 64(N)
//   grid  = (N/64 = 4, M/32)
// A-matrix 16x32 f16 lane layout (ISA 7.12.2): lane l, m=l%16, kg=(l/16)*8:
//   elem i in [0,8)  -> K = k0 + kg + i
//   elem i in [8,16) -> K = k0 + 16 + kg + (i-8)
// B-matrix 32x16 mirrored over lanes: n = l%16, same K order per element ->
//   two contiguous 16B loads from Wt row n.
// C/D 16x16 f32: VGPR r -> row (r + 8*(l/16)), col l%16.
// ---------------------------------------------------------------------------
__global__ __launch_bounds__(256)
void gemm256_wmma_kernel(const float* __restrict__ A,
                         const _Float16* __restrict__ Wt,
                         const float* __restrict__ bias,
                         float* __restrict__ Out,
                         int relu)
{
    const int lane   = threadIdx.x & 31;
    const int wave   = threadIdx.x >> 5;
    const int tile_m = blockIdx.y * 32 + (wave >> 2) * 16;
    const int tile_n = blockIdx.x * 64 + (wave & 3) * 16;

    const int ln16 = lane & 15;         // row (A) / col (B,C,D)
    const int kg   = (lane >> 4) << 3;  // 0 or 8

    const float*    __restrict__ arow = A  + (size_t)(tile_m + ln16) * H_;
    const int n = tile_n + ln16;
    const _Float16* __restrict__ wrow = Wt + (size_t)n * H_;

    v8f acc = {};
    #pragma unroll
    for (int k0 = 0; k0 < H_; k0 += 32) {
        // A: 16 f32 -> f16 (four 16B loads, 32B-aligned since kg in {0,8})
        float4 a0 = *(const float4*)(arow + k0 + kg);
        float4 a1 = *(const float4*)(arow + k0 + kg + 4);
        float4 a2 = *(const float4*)(arow + k0 + kg + 16);
        float4 a3 = *(const float4*)(arow + k0 + kg + 20);
        v16h a;
        a[0]  = (_Float16)a0.x; a[1]  = (_Float16)a0.y;
        a[2]  = (_Float16)a0.z; a[3]  = (_Float16)a0.w;
        a[4]  = (_Float16)a1.x; a[5]  = (_Float16)a1.y;
        a[6]  = (_Float16)a1.z; a[7]  = (_Float16)a1.w;
        a[8]  = (_Float16)a2.x; a[9]  = (_Float16)a2.y;
        a[10] = (_Float16)a2.z; a[11] = (_Float16)a2.w;
        a[12] = (_Float16)a3.x; a[13] = (_Float16)a3.y;
        a[14] = (_Float16)a3.z; a[15] = (_Float16)a3.w;

        // B: two contiguous 16B f16 loads from the transposed weight row
        v8h b0 = *(const v8h*)(wrow + k0 + kg);
        v8h b1 = *(const v8h*)(wrow + k0 + 16 + kg);
        v16h b = __builtin_shufflevector(b0, b1,
                     0, 1, 2, 3, 4, 5, 6, 7, 8, 9, 10, 11, 12, 13, 14, 15);

        acc = __builtin_amdgcn_wmma_f32_16x16x32_f16(
                  /*neg_a=*/false, a, /*neg_b=*/false, b,
                  /*c_mod=*/(short)0, acc, /*reuse_a=*/false, /*reuse_b=*/false);
    }

    const int rbase = tile_m + ((lane >> 4) << 3);
    const float bn = bias[n];
    #pragma unroll
    for (int r = 0; r < 8; ++r) {
        float v = acc[r] + bn;
        if (relu) v = fmaxf(v, 0.f);
        Out[(size_t)(rbase + r) * H_ + n] = v;
    }
}

// ---------------------------------------------------------------------------
// Edge attention + scatter.  One wave32 per (b, e, head): lane = head-dim d.
//   scores_f = <q, k_f>/sqrt(32)  (butterfly shfl_xor reduce)
//   softmax over F=16, attn_d = sum_f w_f * v_f[d]
//   atomicAdd into upd[b, q_idx, d*8 + h]   (torch permute(1,0) channel order)
// ---------------------------------------------------------------------------
__global__ __launch_bounds__(256)
void edge_attn_kernel(const float* __restrict__ Qp,
                      const float* __restrict__ Kp,
                      const float* __restrict__ Vp,
                      const int*   __restrict__ q_idx,
                      const int*   __restrict__ k_idx,
                      float*       __restrict__ upd)
{
    const int lane = threadIdx.x & 31;
    const int wave = threadIdx.x >> 5;
    const int task = blockIdx.x * 8 + wave;    // B*E*NH tasks
    const int h    = task & (NH_ - 1);
    const int be   = task >> 3;                // b*E + e
    const int b    = be >> 12;                 // E = 4096

    const int qi = q_idx[be];
    const float qv = Qp[((size_t)(b * Q_ + qi) * H_) + h * HD_ + lane];

    const int* __restrict__ kptr = k_idx + (size_t)be * F_;
    int   ki[F_];
    float sc[F_];
    float mx = -1e30f;
    #pragma unroll
    for (int f = 0; f < F_; ++f) {
        ki[f] = kptr[f];
        const size_t rowoff = ((size_t)(b * K_ + ki[f]) * H_) + h * HD_ + lane;
        // prefetch the V row we will gather in the second pass
        __builtin_prefetch(&Vp[rowoff], 0, 3);
        float kv = Kp[rowoff];
        float s = qv * kv;
        #pragma unroll
        for (int off = 16; off > 0; off >>= 1) s += __shfl_xor(s, off, 32);
        s *= 0.17677669529663687f;             // 1/sqrt(32)
        sc[f] = s;
        mx = fmaxf(mx, s);
    }

    float denom = 0.f;
    #pragma unroll
    for (int f = 0; f < F_; ++f) { sc[f] = __expf(sc[f] - mx); denom += sc[f]; }
    const float inv = 1.f / denom;

    float attn = 0.f;
    #pragma unroll
    for (int f = 0; f < F_; ++f) {
        float vv = Vp[((size_t)(b * K_ + ki[f]) * H_) + h * HD_ + lane];
        attn += sc[f] * inv * vv;
    }

    atomicAdd(&upd[((size_t)(b * Q_ + qi) * H_) + lane * NH_ + h], attn);
}

// ---------------------------------------------------------------------------
// out = LayerNorm(x + yscale*y) * g + be   (one wave32 per 256-wide row)
// ---------------------------------------------------------------------------
__global__ __launch_bounds__(256)
void add_ln_kernel(const float* __restrict__ x,
                   const float* __restrict__ y,
                   float yscale,
                   const float* __restrict__ g,
                   const float* __restrict__ be,
                   float* __restrict__ out,
                   int rows)
{
    const int lane = threadIdx.x & 31;
    const int wave = threadIdx.x >> 5;
    const int row  = blockIdx.x * 8 + wave;
    if (row >= rows) return;

    const float* __restrict__ xr = x + (size_t)row * H_;
    const float* __restrict__ yr = y + (size_t)row * H_;

    float v[8];
    float s = 0.f, ss = 0.f;
    #pragma unroll
    for (int i = 0; i < 8; ++i) {
        const int c = i * 32 + lane;
        float t = xr[c] + yscale * yr[c];
        v[i] = t;
        s  += t;
        ss += t * t;
    }
    #pragma unroll
    for (int off = 16; off > 0; off >>= 1) {
        s  += __shfl_xor(s,  off, 32);
        ss += __shfl_xor(ss, off, 32);
    }
    const float mean = s * (1.f / H_);
    const float var  = ss * (1.f / H_) - mean * mean;
    const float rstd = rsqrtf(var + 1e-5f);

    #pragma unroll
    for (int i = 0; i < 8; ++i) {
        const int c = i * 32 + lane;
        out[(size_t)row * H_ + c] = (v[i] - mean) * rstd * g[c] + be[c];
    }
}

// ---------------------------------------------------------------------------
// Launch
// ---------------------------------------------------------------------------
extern "C" void kernel_launch(void* const* d_in, const int* in_sizes, int n_in,
                              void* d_out, int out_size, void* d_ws, size_t ws_size,
                              hipStream_t stream)
{
    const float* q_nodes = (const float*)d_in[0];   // [B,Q,H]
    const float* k_nodes = (const float*)d_in[1];   // [B,K,H]
    const int*   q_idx   = (const int*)d_in[2];     // [B,E]
    const int*   k_idx   = (const int*)d_in[3];     // [B,E,F]
    const float* Wq = (const float*)d_in[4];
    const float* Wk = (const float*)d_in[5];
    const float* Wv = (const float*)d_in[6];
    const float* W1 = (const float*)d_in[7];
    const float* W2 = (const float*)d_in[8];
    const float* bq = (const float*)d_in[9];
    const float* bk = (const float*)d_in[10];
    const float* bv = (const float*)d_in[11];
    const float* b1 = (const float*)d_in[12];
    const float* b2 = (const float*)d_in[13];
    const float* g1  = (const float*)d_in[14];
    const float* be1 = (const float*)d_in[15];
    const float* g2  = (const float*)d_in[16];
    const float* be2 = (const float*)d_in[17];
    float* out = (float*)d_out;

    // Workspace: 7 f32 buffers of M_*H_ (16 MB each) + 5 f16 weight buffers
    const size_t NBUF = (size_t)M_ * H_;
    float* Qp  = (float*)d_ws;
    float* Kp  = Qp  + NBUF;
    float* Vp  = Kp  + NBUF;
    float* upd = Vp  + NBUF;
    float* ln1 = upd + NBUF;
    float* h1  = ln1 + NBUF;
    float* ffn = h1  + NBUF;
    _Float16* Wqt = (_Float16*)(ffn + NBUF);
    _Float16* Wkt = Wqt + (size_t)H_ * H_;
    _Float16* Wvt = Wkt + (size_t)H_ * H_;
    _Float16* W1t = Wvt + (size_t)H_ * H_;
    _Float16* W2t = W1t + (size_t)H_ * H_;

    const dim3 blk(256);

    // 0) zero the scatter accumulator; transpose+convert the 5 weights
    {
        int n4 = (int)(NBUF / 4);
        zero_f32_kernel<<<dim3((n4 + 255) / 256), blk, 0, stream>>>(upd, n4);
    }
    const dim3 tgrid(H_ * H_ / 256);
    transpose_cvt_kernel<<<tgrid, blk, 0, stream>>>(Wq, Wqt);
    transpose_cvt_kernel<<<tgrid, blk, 0, stream>>>(Wk, Wkt);
    transpose_cvt_kernel<<<tgrid, blk, 0, stream>>>(Wv, Wvt);
    transpose_cvt_kernel<<<tgrid, blk, 0, stream>>>(W1, W1t);
    transpose_cvt_kernel<<<tgrid, blk, 0, stream>>>(W2, W2t);

    // 1) projections: Qp/Kp/Vp  (M=16384, N=K=256)
    const dim3 ggrid(H_ / 64, M_ / 32);
    gemm256_wmma_kernel<<<ggrid, blk, 0, stream>>>(q_nodes, Wqt, bq, Qp, 0);
    gemm256_wmma_kernel<<<ggrid, blk, 0, stream>>>(k_nodes, Wkt, bk, Kp, 0);
    gemm256_wmma_kernel<<<ggrid, blk, 0, stream>>>(k_nodes, Wvt, bv, Vp, 0);

    // 2) edge attention + scatter-add (B*E*NH waves, 8 waves per block)
    edge_attn_kernel<<<dim3(B_ * E_), blk, 0, stream>>>(Qp, Kp, Vp, q_idx, k_idx, upd);

    // 3) out = LN(q_nodes + upd/(E+1e-6))
    const float escale = 1.0f / ((float)E_ + 1e-6f);
    add_ln_kernel<<<dim3(M_ / 8), blk, 0, stream>>>(q_nodes, upd, escale, g1, be1, ln1, M_);

    // 4) FFN
    gemm256_wmma_kernel<<<ggrid, blk, 0, stream>>>(ln1, W1t, b1, h1, 1);
    gemm256_wmma_kernel<<<ggrid, blk, 0, stream>>>(h1, W2t, b2, ffn, 0);

    // 5) out = LN(ln1 + ffn)
    add_ln_kernel<<<dim3(M_ / 8), blk, 0, stream>>>(ln1, ffn, 1.0f, g2, be2, out, M_);
}
